// Ernie4DecoderLayer_66881230733996
// MI455X (gfx1250) — compile-verified, matrix operand order checked by reading, not run on hardware
//
#include <hip/hip_runtime.h>
#include <hip/hip_bf16.h>

// ---------------- problem constants ----------------
#define Bb    2
#define Ss    1024
#define Hh    1024
#define NHh   16
#define NKVv  4
#define Dd    64
#define Ee    64
#define TOPK  6
#define Ii    512
#define SIi   1024
#define Mm    (Bb*Ss)                       // 2048 tokens
#define QKVN  (NHh*Dd + 2*NKVv*Dd)          // 1536
#define EPSf  1e-5f
#define THETAf 500000.0f
#define SCALEf 0.125f                       // 64^-0.5
#define LDP   33                            // padded LDS stride (bank-conflict free)

typedef __bf16 bf16;
typedef __attribute__((ext_vector_type(16))) __bf16 v16bf;
typedef __attribute__((ext_vector_type(8)))  float  v8f;

// ---------------- WMMA helpers ----------------
__device__ __forceinline__ v8f wmma_bf(v16bf a, v16bf b, v8f c) {
  return __builtin_amdgcn_wmma_f32_16x16x32_bf16(false, a, false, b, (short)0, c, false, false);
}

__device__ __forceinline__ v8f zero8() {
  v8f z;
#pragma unroll
  for (int r = 0; r < 8; ++r) z[r] = 0.0f;
  return z;
}

// A-fragment (16x32 bf16): caller passes rowp = base + (lane&15)*ld + k0 (contiguous k)
__device__ __forceinline__ v16bf afrag_bf16(const bf16* __restrict__ rowp, int lane) {
  int hf = (lane >> 4) & 1;
  v16bf f;
#pragma unroll
  for (int v = 0; v < 8; ++v) {
    int k = ((v >> 2) << 4) | (hf << 3) | ((v & 3) << 1);
    f[2*v]   = rowp[k];
    f[2*v+1] = rowp[k+1];
  }
  return f;
}

// Same pattern but from fp32 source, converting to bf16 on the fly (N,K weight rows)
__device__ __forceinline__ v16bf afrag_f32(const float* __restrict__ rowp, int lane) {
  int hf = (lane >> 4) & 1;
  v16bf f;
#pragma unroll
  for (int v = 0; v < 8; ++v) {
    int k = ((v >> 2) << 4) | (hf << 3) | ((v & 3) << 1);
    f[2*v]   = (bf16)rowp[k];
    f[2*v+1] = (bf16)rowp[k+1];
  }
  return f;
}

// B-fragment from LDS-staged bf16 tile in N-major [n][k] layout, stride LDP
__device__ __forceinline__ v16bf bfrag_lds(const bf16* __restrict__ tile, int n0sub, int lane) {
  int hf = (lane >> 4) & 1;
  int n  = lane & 15;
  const bf16* rowp = tile + (size_t)(n0sub + n) * LDP;
  v16bf f;
#pragma unroll
  for (int v = 0; v < 8; ++v) {
    int k = ((v >> 2) << 4) | (hf << 3) | ((v & 3) << 1);
    f[2*v]   = rowp[k];
    f[2*v+1] = rowp[k+1];
  }
  return f;
}

// ---------------- elementwise / norm kernels ----------------
__global__ __launch_bounds__(256) void rmsnorm_kernel(const float* __restrict__ x,
                                                      const float* __restrict__ w,
                                                      bf16* __restrict__ out) {
  int row = blockIdx.x, tid = threadIdx.x;
  const float* xr = x + (size_t)row * Hh;
  float v[4], ss = 0.f;
#pragma unroll
  for (int i = 0; i < 4; ++i) { v[i] = xr[tid + i*256]; ss += v[i]*v[i]; }
  __shared__ float red[256];
  red[tid] = ss; __syncthreads();
  for (int s = 128; s > 0; s >>= 1) { if (tid < s) red[tid] += red[tid+s]; __syncthreads(); }
  float rstd = rsqrtf(red[0] * (1.0f/Hh) + EPSf);
#pragma unroll
  for (int i = 0; i < 4; ++i) {
    int h = tid + i*256;
    out[(size_t)row*Hh + h] = (bf16)(v[i] * rstd * w[h]);
  }
}

__global__ __launch_bounds__(256) void resid_rms_kernel(const float* __restrict__ attn,
                                                        const float* __restrict__ hidden,
                                                        const float* __restrict__ w,
                                                        float* __restrict__ residual,
                                                        bf16* __restrict__ x2) {
  int row = blockIdx.x, tid = threadIdx.x;
  const float* ar = attn + (size_t)row * Hh;
  const float* hr = hidden + (size_t)row * Hh;
  float v[4], ss = 0.f;
#pragma unroll
  for (int i = 0; i < 4; ++i) {
    int h = tid + i*256;
    v[i] = ar[h] + hr[h];
    residual[(size_t)row*Hh + h] = v[i];
    ss += v[i]*v[i];
  }
  __shared__ float red[256];
  red[tid] = ss; __syncthreads();
  for (int s = 128; s > 0; s >>= 1) { if (tid < s) red[tid] += red[tid+s]; __syncthreads(); }
  float rstd = rsqrtf(red[0] * (1.0f/Hh) + EPSf);
#pragma unroll
  for (int i = 0; i < 4; ++i) {
    int h = tid + i*256;
    x2[(size_t)row*Hh + h] = (bf16)(v[i] * rstd * w[h]);
  }
}

// ---------------- generic GEMM: C[M,N] = Abf16[M,K] @ Wf32[N,K]^T ----------------
// grid: (M/128, N/64), block 256 (8 waves, 4x2), wave tile 32x32
__global__ __launch_bounds__(256) void gemm_wt_kernel(const bf16* __restrict__ A,
                                                      const float* __restrict__ Bw,
                                                      float* __restrict__ C,
                                                      int Kdim, int Ndim) {
  int tid = threadIdx.x, wv = tid >> 5, lane = tid & 31;
  int hf = lane >> 4, ln = lane & 15;
  int wm = wv & 3, wn = wv >> 2;
  int m0 = blockIdx.x * 128 + wm * 32;
  int n0 = blockIdx.y * 64  + wn * 32;
  v8f acc[2][2];
#pragma unroll
  for (int i = 0; i < 2; ++i)
#pragma unroll
    for (int j = 0; j < 2; ++j) acc[i][j] = zero8();
  const bf16*  a0p = A  + (size_t)(m0 + ln)      * Kdim;
  const bf16*  a1p = A  + (size_t)(m0 + 16 + ln) * Kdim;
  const float* b0p = Bw + (size_t)(n0 + ln)      * Kdim;
  const float* b1p = Bw + (size_t)(n0 + 16 + ln) * Kdim;
  for (int k = 0; k < Kdim; k += 32) {
    if (k + 64 < Kdim) {                   // prefetch 2 k-tiles ahead (speculative)
      __builtin_prefetch(b0p + k + 64, 0, 0);
      __builtin_prefetch(b1p + k + 64, 0, 0);
      __builtin_prefetch(a0p + k + 64, 0, 0);
    }
    v16bf a0 = afrag_bf16(a0p + k, lane);
    v16bf a1 = afrag_bf16(a1p + k, lane);
    v16bf b0 = afrag_f32(b0p + k, lane);
    v16bf b1 = afrag_f32(b1p + k, lane);
    acc[0][0] = wmma_bf(a0, b0, acc[0][0]);
    acc[0][1] = wmma_bf(a0, b1, acc[0][1]);
    acc[1][0] = wmma_bf(a1, b0, acc[1][0]);
    acc[1][1] = wmma_bf(a1, b1, acc[1][1]);
  }
#pragma unroll
  for (int i = 0; i < 2; ++i)
#pragma unroll
    for (int j = 0; j < 2; ++j)
#pragma unroll
      for (int r = 0; r < 8; ++r)
        C[(size_t)(m0 + i*16 + hf*8 + r) * Ndim + n0 + j*16 + ln] = acc[i][j][r];
}

// ---------------- RoPE + cache layout ----------------
__global__ __launch_bounds__(128) void rope_kernel(const float* __restrict__ qkv,
                                                   const int* __restrict__ positions,
                                                   bf16* __restrict__ Qbf,
                                                   bf16* __restrict__ Kbf,
                                                   bf16* __restrict__ Vt) {
  int t = blockIdx.x, tid = threadIdx.x;
  int b = t / Ss, s = t % Ss;
  float pos = (float)positions[t];
  const float* row = qkv + (size_t)t * QKVN;
  for (int p = tid; p < NHh * 32; p += 128) {
    int h = p >> 5, i = p & 31;
    float fr = pos * __powf(THETAf, -(float)i * (1.0f/32.0f));
    float c = __cosf(fr), sn = __sinf(fr);
    float x1 = row[h*64 + 2*i], x2 = row[h*64 + 2*i + 1];
    size_t base = (((size_t)b*NHh + h)*Ss + s)*Dd;
    Qbf[base + 2*i]     = (bf16)(x1*c - x2*sn);
    Qbf[base + 2*i + 1] = (bf16)(x2*c + x1*sn);
  }
  for (int p = tid; p < NKVv * 32; p += 128) {
    int kh = p >> 5, i = p & 31;
    float fr = pos * __powf(THETAf, -(float)i * (1.0f/32.0f));
    float c = __cosf(fr), sn = __sinf(fr);
    float x1 = row[NHh*Dd + kh*64 + 2*i], x2 = row[NHh*Dd + kh*64 + 2*i + 1];
    size_t base = (((size_t)b*NKVv + kh)*Ss + s)*Dd;
    Kbf[base + 2*i]     = (bf16)(x1*c - x2*sn);
    Kbf[base + 2*i + 1] = (bf16)(x2*c + x1*sn);
  }
  for (int e = tid; e < NKVv * Dd; e += 128) {
    int kh = e >> 6, d = e & 63;
    Vt[(((size_t)b*NKVv + kh)*Dd + d)*Ss + s] = (bf16)row[NHh*Dd + NKVv*Dd + kh*64 + d];
  }
}

// ---------------- flash attention: online softmax, all-WMMA ----------------
__global__ __launch_bounds__(256) void flash_kernel(const bf16* __restrict__ Qbf,
                                                    const bf16* __restrict__ Kbf,
                                                    const bf16* __restrict__ Vt,
                                                    bf16* __restrict__ ctx) {
  int bid = blockIdx.x;
  int qb = bid & 7, h = (bid >> 3) & 15, b = bid >> 7;
  int tid = threadIdx.x, wv = tid >> 5, lane = tid & 31;
  int hf = lane >> 4, ln = lane & 15;
  int kvh = h >> 2;                    // NH/NKV = 4
  int q0 = qb * 128 + wv * 16;
  const bf16* Qbase = Qbf + (((size_t)b*NHh + h)*Ss + q0)*Dd + (size_t)ln*Dd;
  const bf16* Kbase = Kbf + (((size_t)b*NKVv + kvh)*Ss)*Dd;
  const bf16* Vbase = Vt  + (((size_t)b*NKVv + kvh)*Dd)*Ss;
  v16bf qa0 = afrag_bf16(Qbase,      lane);      // d 0..31
  v16bf qa1 = afrag_bf16(Qbase + 32, lane);      // d 32..63
  v8f acc[4];
#pragma unroll
  for (int c = 0; c < 4; ++c) acc[c] = zero8();
  float mrow[8], lrow[8];
#pragma unroll
  for (int r = 0; r < 8; ++r) { mrow[r] = -1e38f; lrow[r] = 0.f; }
  __shared__ bf16 pbuf[8][16*32];
  bf16* pb = pbuf[wv];

  for (int kt = 0; kt < q0 + 16; kt += 32) {
    v8f s0 = zero8(), s1 = zero8();
    s0 = wmma_bf(qa0, afrag_bf16(Kbase + (size_t)kt*Dd + (size_t)ln*Dd,        lane), s0);
    s0 = wmma_bf(qa1, afrag_bf16(Kbase + (size_t)kt*Dd + (size_t)ln*Dd + 32,   lane), s0);
    s1 = wmma_bf(qa0, afrag_bf16(Kbase + (size_t)(kt+16)*Dd + (size_t)ln*Dd,   lane), s1);
    s1 = wmma_bf(qa1, afrag_bf16(Kbase + (size_t)(kt+16)*Dd + (size_t)ln*Dd+32,lane), s1);
    bool needmask = (kt + 31) > q0;
#pragma unroll
    for (int r = 0; r < 8; ++r) {
      float sv0 = s0[r] * SCALEf, sv1 = s1[r] * SCALEf;
      if (needmask) {
        int rowg = q0 + hf*8 + r;
        if (kt + ln      > rowg) sv0 = -1e30f;
        if (kt + 16 + ln > rowg) sv1 = -1e30f;
      }
      float mx = fmaxf(sv0, sv1);
#pragma unroll
      for (int msk = 8; msk >= 1; msk >>= 1) mx = fmaxf(mx, __shfl_xor(mx, msk, 32));
      float newm = fmaxf(mrow[r], mx);
      float p0 = __expf(sv0 - newm), p1 = __expf(sv1 - newm);
      float alpha = __expf(mrow[r] - newm);
      mrow[r] = newm;
      float rs = p0 + p1;
#pragma unroll
      for (int msk = 8; msk >= 1; msk >>= 1) rs += __shfl_xor(rs, msk, 32);
      lrow[r] = lrow[r] * alpha + rs;
#pragma unroll
      for (int c = 0; c < 4; ++c) acc[c][r] *= alpha;
      pb[(hf*8 + r)*32 + ln]      = (bf16)p0;
      pb[(hf*8 + r)*32 + ln + 16] = (bf16)p1;
    }
    v16bf pf = afrag_bf16(pb + (size_t)ln*32, lane);
#pragma unroll
    for (int c = 0; c < 4; ++c)
      acc[c] = wmma_bf(pf, afrag_bf16(Vbase + (size_t)(c*16 + ln)*Ss + kt, lane), acc[c]);
  }
#pragma unroll
  for (int r = 0; r < 8; ++r) {
    float inv = 1.0f / lrow[r];
    int rowg = q0 + hf*8 + r;
    size_t ob = ((size_t)b*Ss + rowg)*(NHh*Dd) + (size_t)h*Dd;
#pragma unroll
    for (int c = 0; c < 4; ++c)
      ctx[ob + c*16 + ln] = (bf16)(acc[c][r] * inv);
  }
}

// ---------------- fused gate/up SwiGLU GEMM (shared expert + MoE experts) ----------------
// gather mode (list!=null): grid.x = expert, loops its row tiles; direct: grid.x = row tile
// block 256 = 8 waves, each wave: 16 rows x 32 cols; weight tiles staged in LDS once/block
__global__ __launch_bounds__(256) void gateup_kernel(const bf16* __restrict__ X,
                                                     const float* __restrict__ Wg,
                                                     const float* __restrict__ Wu,
                                                     bf16* __restrict__ Act,
                                                     const int* __restrict__ list,
                                                     const int* __restrict__ counts,
                                                     const int* __restrict__ offsets,
                                                     int Kdim, int Ndim) {
  int tid = threadIdx.x, wv = tid >> 5, lane = tid & 31;
  int hf = lane >> 4, ln = lane & 15;
  int n0 = blockIdx.y * 32;
  int cnt, off;
  const float *wg, *wu;
  if (list) {
    int e = blockIdx.x;
    cnt = counts[e]; off = offsets[e];
    wg = Wg + (size_t)e * Kdim * Ndim;
    wu = Wu + (size_t)e * Kdim * Ndim;
    if (cnt == 0) return;
  } else {
    cnt = Mm; off = 0; wg = Wg; wu = Wu;
  }
  __shared__ bf16 ldsG[32 * LDP];            // [n][k], padded
  __shared__ bf16 ldsU[32 * LDP];
  int kk  = tid >> 3;                        // 0..31 : k row staged by this thread
  int nn4 = (tid & 7) * 4;                   // 0..28 : 4 consecutive n

  int tb    = list ? 0   : blockIdx.x * 128;
  int tbLim = list ? cnt : tb + 1;
  for (; tb < tbLim; tb += 128) {
    int ridx = tb + wv*16 + ln;
    int rc = ridx < cnt - 1 ? ridx : cnt - 1;
    int rowg = list ? list[off + rc] : ridx;
    const bf16* arow = X + (size_t)rowg * Kdim;
    v8f ag[2], au[2];
#pragma unroll
    for (int j = 0; j < 2; ++j) { ag[j] = zero8(); au[j] = zero8(); }
    for (int k = 0; k < Kdim; k += 32) {
      // cooperative coalesced stage: 32k x 32n fp32 -> bf16, transposed to [n][k]
      {
        const float* gs = wg + (size_t)(k + kk) * Ndim + n0 + nn4;
        const float* us = wu + (size_t)(k + kk) * Ndim + n0 + nn4;
        if (k + 64 < Kdim) {
          __builtin_prefetch(gs + (size_t)64 * Ndim, 0, 0);
          __builtin_prefetch(us + (size_t)64 * Ndim, 0, 0);
        }
        float4 gv = *(const float4*)gs;
        float4 uv = *(const float4*)us;
#pragma unroll
        for (int i = 0; i < 4; ++i) {
          ldsG[(nn4 + i) * LDP + kk] = (bf16)((&gv.x)[i]);
          ldsU[(nn4 + i) * LDP + kk] = (bf16)((&uv.x)[i]);
        }
      }
      __syncthreads();
      v16bf a   = afrag_bf16(arow + k, lane);
      v16bf bg0 = bfrag_lds(ldsG, 0,  lane);
      v16bf bg1 = bfrag_lds(ldsG, 16, lane);
      v16bf bu0 = bfrag_lds(ldsU, 0,  lane);
      v16bf bu1 = bfrag_lds(ldsU, 16, lane);
      ag[0] = wmma_bf(a, bg0, ag[0]);
      ag[1] = wmma_bf(a, bg1, ag[1]);
      au[0] = wmma_bf(a, bu0, au[0]);
      au[1] = wmma_bf(a, bu1, au[1]);
      __syncthreads();
    }
#pragma unroll
    for (int j = 0; j < 2; ++j)
#pragma unroll
      for (int r = 0; r < 8; ++r) {
        int rl = tb + wv*16 + hf*8 + r;
        if (rl < cnt) {
          float g = ag[j][r], u = au[j][r];
          float act = g / (1.0f + __expf(-g)) * u;   // silu(g)*u
          Act[(size_t)(off + rl) * Ndim + n0 + j*16 + ln] = (bf16)act;
        }
      }
  }
}

// ---------------- down projection (shared: plain store; experts: scaled atomic add) ----------
// block tile 128x64, wave 32x32 (4x2 waves); weight tile staged in LDS once/block
__global__ __launch_bounds__(256) void down_kernel(const bf16* __restrict__ Act,
                                                   const float* __restrict__ Wd,
                                                   float* __restrict__ Y,
                                                   const int* __restrict__ list,
                                                   const float* __restrict__ listw,
                                                   const int* __restrict__ counts,
                                                   const int* __restrict__ offsets,
                                                   int Kdim, int Ndim) {
  int tid = threadIdx.x, wv = tid >> 5, lane = tid & 31;
  int hf = lane >> 4, ln = lane & 15;
  int wm = wv & 3, wn = wv >> 2;
  int n0b = blockIdx.y * 64;                 // block's 64-wide N window
  int cnt, off;
  const float* wd;
  if (list) {
    int e = blockIdx.x;
    cnt = counts[e]; off = offsets[e];
    wd = Wd + (size_t)e * Kdim * Ndim;
    if (cnt == 0) return;
  } else {
    cnt = Mm; off = 0; wd = Wd;
  }
  __shared__ bf16 ldsD[64 * LDP];            // [n][k], padded
  int kk  = tid >> 3;
  int nn4 = (tid & 7) * 4;

  int tb    = list ? 0   : blockIdx.x * 128;
  int tbLim = list ? cnt : tb + 1;
  for (; tb < tbLim; tb += 128) {
    int m0 = tb + wm * 32;
    int r0 = m0 + ln;       int r0c = r0 < cnt - 1 ? r0 : cnt - 1;
    int r1 = m0 + 16 + ln;  int r1c = r1 < cnt - 1 ? r1 : cnt - 1;
    const bf16* a0p = Act + (size_t)(off + r0c) * Kdim;
    const bf16* a1p = Act + (size_t)(off + r1c) * Kdim;
    v8f acc[2][2];
#pragma unroll
    for (int i = 0; i < 2; ++i)
#pragma unroll
      for (int j = 0; j < 2; ++j) acc[i][j] = zero8();
    for (int k = 0; k < Kdim; k += 32) {
      {
        const float* ds0 = wd + (size_t)(k + kk) * Ndim + n0b + nn4;
        if (k + 64 < Kdim) __builtin_prefetch(ds0 + (size_t)64 * Ndim, 0, 0);
        float4 v0 = *(const float4*)ds0;
        float4 v1 = *(const float4*)(ds0 + 32);
#pragma unroll
        for (int i = 0; i < 4; ++i) {
          ldsD[(nn4 + i)      * LDP + kk] = (bf16)((&v0.x)[i]);
          ldsD[(nn4 + 32 + i) * LDP + kk] = (bf16)((&v1.x)[i]);
        }
      }
      __syncthreads();
      v16bf a0 = afrag_bf16(a0p + k, lane);
      v16bf a1 = afrag_bf16(a1p + k, lane);
      v16bf b0 = bfrag_lds(ldsD, wn*32,      lane);
      v16bf b1 = bfrag_lds(ldsD, wn*32 + 16, lane);
      acc[0][0] = wmma_bf(a0, b0, acc[0][0]);
      acc[0][1] = wmma_bf(a0, b1, acc[0][1]);
      acc[1][0] = wmma_bf(a1, b0, acc[1][0]);
      acc[1][1] = wmma_bf(a1, b1, acc[1][1]);
      __syncthreads();
    }
    int n0 = n0b + wn * 32;
#pragma unroll
    for (int i = 0; i < 2; ++i)
#pragma unroll
      for (int j = 0; j < 2; ++j)
#pragma unroll
        for (int r = 0; r < 8; ++r) {
          int rl = m0 + i*16 + hf*8 + r;
          int col = n0 + j*16 + ln;
          if (list) {
            if (rl < cnt) {
              int tok = list[off + rl];
              float wgt = listw[off + rl];
              atomicAdd(&Y[(size_t)tok * Ndim + col], acc[i][j][r] * wgt);
            }
          } else {
            Y[(size_t)rl * Ndim + col] = acc[i][j][r];
          }
        }
  }
}

// ---------------- router: sigmoid + biased top-6, counts ----------------
__global__ void router_kernel(const bf16* __restrict__ x2,
                              const float* __restrict__ gate_w,
                              const float* __restrict__ gate_bias,
                              int* __restrict__ topi, float* __restrict__ tw,
                              int* __restrict__ counts) {
  int t = blockIdx.x, e = threadIdx.x;   // 64 threads
  const bf16* xr = x2 + (size_t)t * Hh;
  const float* wr = gate_w + (size_t)e * Hh;
  float d = 0.f;
  for (int k = 0; k < Hh; ++k) d += (float)xr[k] * wr[k];
  __shared__ float sig[Ee], sb[Ee];
  float sg = 1.0f / (1.0f + __expf(-d));
  sig[e] = sg; sb[e] = sg + gate_bias[e];
  __syncthreads();
  if (e == 0) {
    unsigned long long taken = 0ull;
    int idx[TOPK]; float wv6[TOPK]; float wsum = 0.f;
    for (int kk = 0; kk < TOPK; ++kk) {
      float bv = -1e30f; int bi = 0;
      for (int j = 0; j < Ee; ++j)
        if (!((taken >> j) & 1ull) && sb[j] > bv) { bv = sb[j]; bi = j; }
      taken |= 1ull << bi;
      idx[kk] = bi; wv6[kk] = sig[bi]; wsum += sig[bi];
    }
    float inv = 1.0f / wsum;
    for (int kk = 0; kk < TOPK; ++kk) {
      topi[t*TOPK + kk] = idx[kk];
      tw[t*TOPK + kk] = wv6[kk] * inv;
      atomicAdd(&counts[idx[kk]], 1);
    }
  }
}

__global__ void zero_counts_kernel(int* __restrict__ c) {
  if (threadIdx.x < Ee) c[threadIdx.x] = 0;
}

__global__ void scan_kernel(const int* __restrict__ counts, int* __restrict__ offsets,
                            int* __restrict__ cursor) {
  if (threadIdx.x == 0) {
    int acc = 0;
    for (int e = 0; e < Ee; ++e) { offsets[e] = acc; cursor[e] = acc; acc += counts[e]; }
  }
}

__global__ void fill_kernel(const int* __restrict__ topi, const float* __restrict__ tw,
                            int* __restrict__ cursor, int* __restrict__ list,
                            float* __restrict__ listw) {
  int t = blockIdx.x * blockDim.x + threadIdx.x;
  if (t >= Mm) return;
  for (int kk = 0; kk < TOPK; ++kk) {
    int e = topi[t*TOPK + kk];
    int pos = atomicAdd(&cursor[e], 1);
    list[pos]  = t;
    listw[pos] = tw[t*TOPK + kk];
  }
}

// ---------------- host-side orchestration ----------------
extern "C" void kernel_launch(void* const* d_in, const int* in_sizes, int n_in,
                              void* d_out, int out_size, void* d_ws, size_t ws_size,
                              hipStream_t stream) {
  const int*   positions = (const int*)  d_in[0];
  const float* hidden    = (const float*)d_in[1];
  const float* ln1_w     = (const float*)d_in[2];
  const float* ln2_w     = (const float*)d_in[3];
  const float* qkv_w     = (const float*)d_in[4];
  const float* o_w       = (const float*)d_in[5];
  const float* gate_w    = (const float*)d_in[6];
  const float* gate_bias = (const float*)d_in[7];
  const float* w_gate    = (const float*)d_in[8];
  const float* w_up      = (const float*)d_in[9];
  const float* w_down    = (const float*)d_in[10];
  const float* ws_gate   = (const float*)d_in[11];
  const float* ws_up     = (const float*)d_in[12];
  const float* ws_down   = (const float*)d_in[13];

  float* y        = (float*)d_out;
  float* residual = y + (size_t)Mm * Hh;

  char* wsp = (char*)d_ws;
  size_t off = 0;
  auto alloc = [&](size_t bytes) -> void* {
    void* p = wsp + off;
    off = (off + bytes + 255) & ~(size_t)255;
    return p;
  };
  bf16*  xbf     = (bf16*) alloc((size_t)Mm * Hh * 2);
  float* qkvbuf  = (float*)alloc((size_t)Mm * QKVN * 4);
  bf16*  Qbf     = (bf16*) alloc((size_t)Mm * NHh * Dd * 2);
  bf16*  Kbf     = (bf16*) alloc((size_t)Mm * NKVv * Dd * 2);
  bf16*  Vt      = (bf16*) alloc((size_t)Mm * NKVv * Dd * 2);
  bf16*  ctxbf   = (bf16*) alloc((size_t)Mm * NHh * Dd * 2);
  float* attnout = (float*)alloc((size_t)Mm * Hh * 4);
  bf16*  x2bf    = (bf16*) alloc((size_t)Mm * Hh * 2);
  bf16*  shbuf   = (bf16*) alloc((size_t)Mm * SIi * 2);
  bf16*  actbuf  = (bf16*) alloc((size_t)Mm * TOPK * Ii * 2);
  int*   topi    = (int*)  alloc((size_t)Mm * TOPK * 4);
  float* tw      = (float*)alloc((size_t)Mm * TOPK * 4);
  int*   counts  = (int*)  alloc(Ee * 4);
  int*   offsets = (int*)  alloc(Ee * 4);
  int*   cursor  = (int*)  alloc(Ee * 4);
  int*   list    = (int*)  alloc((size_t)Mm * TOPK * 4);
  float* listw   = (float*)alloc((size_t)Mm * TOPK * 4);

  // 1. pre-attn RMSNorm -> bf16
  rmsnorm_kernel<<<Mm, 256, 0, stream>>>(hidden, ln1_w, xbf);
  // 2. QKV GEMM (2048x1536x1024)
  gemm_wt_kernel<<<dim3(Mm/128, QKVN/64), 256, 0, stream>>>(xbf, qkv_w, qkvbuf, Hh, QKVN);
  // 3. RoPE + KV cache layouts
  rope_kernel<<<Mm, 128, 0, stream>>>(qkvbuf, positions, Qbf, Kbf, Vt);
  // 4. flash attention
  flash_kernel<<<Bb * NHh * (Ss/128), 256, 0, stream>>>(Qbf, Kbf, Vt, ctxbf);
  // 5. output projection
  gemm_wt_kernel<<<dim3(Mm/128, Hh/64), 256, 0, stream>>>(ctxbf, o_w, attnout, NHh*Dd, Hh);
  // 6. residual + post-attn RMSNorm
  resid_rms_kernel<<<Mm, 256, 0, stream>>>(attnout, hidden, ln2_w, residual, x2bf);
  // 7. shared expert gate/up (direct mode)
  gateup_kernel<<<dim3(Mm/128, SIi/32), 256, 0, stream>>>(x2bf, ws_gate, ws_up, shbuf,
                                                          nullptr, nullptr, nullptr, Hh, SIi);
  // 8-11. routing
  zero_counts_kernel<<<1, 64, 0, stream>>>(counts);
  router_kernel<<<Mm, 64, 0, stream>>>(x2bf, gate_w, gate_bias, topi, tw, counts);
  scan_kernel<<<1, 1, 0, stream>>>(counts, offsets, cursor);
  fill_kernel<<<Mm/256, 256, 0, stream>>>(topi, tw, cursor, list, listw);
  // 12. expert gate/up (gather mode)
  gateup_kernel<<<dim3(Ee, Ii/32), 256, 0, stream>>>(x2bf, w_gate, w_up, actbuf,
                                                     list, counts, offsets, Hh, Ii);
  // 13. shared expert down -> initializes y
  down_kernel<<<dim3(Mm/128, Hh/64), 256, 0, stream>>>(shbuf, ws_down, y,
                                                       nullptr, nullptr, nullptr, nullptr,
                                                       SIi, Hh);
  // 14. expert down -> scaled atomic accumulate into y
  down_kernel<<<dim3(Ee, Hh/64), 256, 0, stream>>>(actbuf, w_down, y,
                                                   list, listw, counts, offsets, Ii, Hh);
  (void)in_sizes; (void)n_in; (void)out_size; (void)ws_size;
}
